// GridNet_43241730736142
// MI455X (gfx1250) — matrix operand order repaired and checked by reading, not compiled, not run on hardware
//
#include <hip/hip_runtime.h>
#include <hip/hip_bf16.h>

typedef __attribute__((ext_vector_type(16))) __bf16       v16bf;
typedef __attribute__((ext_vector_type(8)))  float        v8f;
typedef __attribute__((ext_vector_type(8)))  unsigned int v8u;

#define WSTR   136   // weight row stride (ushorts): 272B -> 16B-aligned rows, conflict-free banks
#define ASTR   136   // activation row stride (ushorts)
#define RAWSTR 36    // raw interp row stride (floats): 144B -> 16B-aligned
#define ROWS_PER_BLOCK 64
#define THREADS 128

#if defined(__has_builtin)
#  if __has_builtin(__builtin_amdgcn_cvt_pk_bf16_f32)
#    define HAVE_CVT_PK_BF16 1
#  endif
#endif

// single f32 -> bf16 (RNE) via native cast
__device__ __forceinline__ unsigned short f2bf(float x) {
  return __builtin_bit_cast(unsigned short, (__bf16)x);
}
// two f32 -> packed 2xbf16 in one u32 (uses v_cvt_pk_bf16_f32 when available)
__device__ __forceinline__ unsigned cvt2(float lo, float hi) {
#ifdef HAVE_CVT_PK_BF16
  auto r = __builtin_amdgcn_cvt_pk_bf16_f32(lo, hi);
  return __builtin_bit_cast(unsigned, r);
#else
  return (unsigned)f2bf(lo) | ((unsigned)f2bf(hi) << 16);
#endif
}
__device__ __forceinline__ v8f vzero8() {
  v8f z = {0.f, 0.f, 0.f, 0.f, 0.f, 0.f, 0.f, 0.f};
  return z;
}
__device__ __forceinline__ v8f wmma_bf16(v16bf a, v16bf b, v8f c) {
  // D = A(16x32 bf16) * B(32x16 bf16) + C(16x16 f32)
  return __builtin_amdgcn_wmma_f32_16x16x32_bf16(false, a, false, b, (short)0, c, false, false);
}

// B fragment (32x16 bf16): lane = N (lane&15); K base b = kofs + (hi-lane ? 8 : 0);
// VGPRs 0-3 hold K=b..b+7, VGPRs 4-7 hold K=b+16..b+23.
// Weights stored transposed [out][in]; a lane reads two 16B-aligned 16-byte chunks, 32 bytes apart.
__device__ __forceinline__ v16bf load_b_frag(const unsigned short* Wb, int row_base, int kofs) {
  int lane = threadIdx.x & 31;
  const unsigned short* p = Wb + (row_base + (lane & 15)) * WSTR + kofs + ((lane & 16) ? 8 : 0);
  uint4 q0 = *(const uint4*)(p);
  uint4 q1 = *(const uint4*)(p + 16);
  v8u u;
  u[0] = q0.x; u[1] = q0.y; u[2] = q0.z; u[3] = q0.w;
  u[4] = q1.x; u[5] = q1.y; u[6] = q1.z; u[7] = q1.w;
  return __builtin_bit_cast(v16bf, u);
}

// A fragment (16x32 bf16): lane = M (lane&15); K base = kofs + (hi-lane ? 8 : 0);
// VGPRs 0-3 hold K=b..b+7, VGPRs 4-7 hold K=b+16..b+23. Two aligned 16B chunks.
__device__ __forceinline__ v16bf load_a_frag_bf(const unsigned short* Ab, int row_base, int kofs) {
  int lane = threadIdx.x & 31;
  const unsigned short* p = Ab + (row_base + (lane & 15)) * ASTR + kofs + ((lane & 16) ? 8 : 0);
  uint4 q0 = *(const uint4*)(p);
  uint4 q1 = *(const uint4*)(p + 16);
  v8u u;
  u[0] = q0.x; u[1] = q0.y; u[2] = q0.z; u[3] = q0.w;
  u[4] = q1.x; u[5] = q1.y; u[6] = q1.z; u[7] = q1.w;
  return __builtin_bit_cast(v16bf, u);
}

// A fragment from the fp32 interp staging (K = 32), packed conversion
__device__ __forceinline__ v16bf load_a_frag_raw(const float* Ab, int row_base) {
  int lane = threadIdx.x & 31;
  const float* p = Ab + (row_base + (lane & 15)) * RAWSTR + ((lane & 16) ? 8 : 0);
  float4 f0 = *(const float4*)(p);
  float4 f1 = *(const float4*)(p + 4);
  float4 f2 = *(const float4*)(p + 16);
  float4 f3 = *(const float4*)(p + 20);
  v8u u;
  u[0] = cvt2(f0.x, f0.y); u[1] = cvt2(f0.z, f0.w);
  u[2] = cvt2(f1.x, f1.y); u[3] = cvt2(f1.z, f1.w);
  u[4] = cvt2(f2.x, f2.y); u[5] = cvt2(f2.z, f2.w);
  u[6] = cvt2(f3.x, f3.y); u[7] = cvt2(f3.z, f3.w);
  return __builtin_bit_cast(v16bf, u);
}

// A fragment directly from global nb_params [N,16] fp32, zero-padded to K=32
__device__ __forceinline__ v16bf load_a_frag_nb(const float* A, int row_base, int npts) {
  int lane = threadIdx.x & 31;
  int row = row_base + (lane & 15);
  if (row > npts - 1) row = npts - 1;
  const float* p = A + (size_t)row * 16 + ((lane & 16) ? 8 : 0);
  float4 f0 = *(const float4*)(p);
  float4 f1 = *(const float4*)(p + 4);
  v8u u;
  u[0] = cvt2(f0.x, f0.y); u[1] = cvt2(f0.z, f0.w);
  u[2] = cvt2(f1.x, f1.y); u[3] = cvt2(f1.z, f1.w);
  u[4] = 0u; u[5] = 0u; u[6] = 0u; u[7] = 0u;   // K = 16..31 zero padding
  return __builtin_bit_cast(v16bf, u);
}

// Scatter a D tile (f32 C/D layout: lane = N, VGPR r -> M = r + hi*8) to bf16 staging
__device__ __forceinline__ void store_d_bf(unsigned short* buf, int row_base, int col_base, v8f d) {
  int lane = threadIdx.x & 31;
  int n = lane & 15;
  int mo = (lane & 16) ? 8 : 0;
#pragma unroll
  for (int r = 0; r < 8; ++r)
    buf[(row_base + mo + r) * ASTR + col_base + n] = f2bf(d[r]);
}

__global__ void __launch_bounds__(THREADS, 1)
gridnet_fused(const float* __restrict__ pos,  const float* __restrict__ nb,
              const float* __restrict__ g0,   const float* __restrict__ g1,
              const float* __restrict__ ligW, const float* __restrict__ ligb,
              const float* __restrict__ recW, const float* __restrict__ recb,
              const float* __restrict__ lbW,  const float* __restrict__ lbb,
              const float* __restrict__ rbW,  const float* __restrict__ rbb,
              const float* __restrict__ fW,   const float* __restrict__ fb,
              float* __restrict__ out, int Npts) {
  __shared__ __align__(16) unsigned short sW[256 * WSTR];               // 68 KB weight stage
  __shared__ __align__(16) unsigned short sLig[ROWS_PER_BLOCK * ASTR];  // 17 KB bf16 lig_x
  __shared__ __align__(16) unsigned short sRec[ROWS_PER_BLOCK * ASTR];  // 17 KB bf16 rec_x
  __shared__ __align__(16) unsigned short sBoth[ROWS_PER_BLOCK * ASTR]; // 17 KB bf16 both
  __shared__ __align__(16) float sRaw[ROWS_PER_BLOCK * RAWSTR];         // 9 KB f32 interp result
  __shared__ float sBias[768];

  const int tid  = threadIdx.x;
  const int wave = tid >> 5;
  const int row0 = blockIdx.x * ROWS_PER_BLOCK;
  const int rb   = wave * 16;           // this wave's local row base
  const int lane = tid & 31;
  const int nidx = lane & 15;

  // ---- biases ----
  for (int i = tid; i < 768; i += THREADS) {
    float v;
    if (i < 128)      v = ligb[i];
    else if (i < 256) v = recb[i - 128];
    else if (i < 384) v = lbb[i - 256];
    else if (i < 512) v = rbb[i - 384];
    else              v = fb[i - 512];
    sBias[i] = v;
  }

  // ---- weight phase 1: ligW^T (rows 0..127, K zero-padded 16..31) + recW^T (rows 128..255)
  // adjacent lanes walk the contiguous 'out' dim -> coalesced global reads
  for (int i = tid; i < 128 * 32; i += THREADS) {
    int o = i & 127, k = i >> 7;
    sW[o * WSTR + k]         = (k < 16) ? f2bf(ligW[k * 128 + o]) : (unsigned short)0;
    sW[(128 + o) * WSTR + k] = f2bf(recW[k * 128 + o]);
  }

  // ---- phase 0: tricubic B-spline gather; one (row, grid) per thread ----
  {
    int tl   = tid & 63;
    int gsel = tid >> 6;
    int row = row0 + tl; if (row > Npts - 1) row = Npts - 1;
    const float* __restrict__ G = gsel ? g1 : g0;
    float w[3][4]; int ix[3][4];
#pragma unroll
    for (int d = 0; d < 3; ++d) {
      float p = pos[row * 3 + d];
      float v = 0.5f + p * (0.5f / 16.0f);
      v = fminf(fmaxf(v, 0.f), 1.f);
      float u  = v * 64.0f;
      float fl = floorf(u);
      int   i0 = (int)fl;
      float t  = u - fl;
#pragma unroll
      for (int o = 0; o < 4; ++o) {
        int id = i0 - 1 + o;
        ix[d][o] = (id < 0) ? 0 : (id > 64 ? 64 : id);
      }
      float t2 = t * t, t3 = t2 * t, omt = 1.f - t;
      w[d][0] = omt * omt * omt * (1.f / 6.f);
      w[d][1] = (3.f * t3 - 6.f * t2 + 4.f) * (1.f / 6.f);
      w[d][2] = (-3.f * t3 + 3.f * t2 + 3.f * t + 1.f) * (1.f / 6.f);
      w[d][3] = t3 * (1.f / 6.f);
    }
    float acc[16];
#pragma unroll
    for (int c = 0; c < 16; ++c) acc[c] = 0.f;
    for (int i = 0; i < 4; ++i)
      for (int j = 0; j < 4; ++j) {
        float wij = w[0][i] * w[1][j];
        int   bij = (ix[0][i] * 65 + ix[1][j]) * 65;
#pragma unroll
        for (int k = 0; k < 4; ++k) {
          float www = wij * w[2][k];
          const float4* gp = (const float4*)(G + (size_t)(bij + ix[2][k]) * 16);
          float4 a0 = gp[0], a1 = gp[1], a2 = gp[2], a3 = gp[3];
          acc[0]  += www * a0.x; acc[1]  += www * a0.y; acc[2]  += www * a0.z; acc[3]  += www * a0.w;
          acc[4]  += www * a1.x; acc[5]  += www * a1.y; acc[6]  += www * a1.z; acc[7]  += www * a1.w;
          acc[8]  += www * a2.x; acc[9]  += www * a2.y; acc[10] += www * a2.z; acc[11] += www * a2.w;
          acc[12] += www * a3.x; acc[13] += www * a3.y; acc[14] += www * a3.z; acc[15] += www * a3.w;
        }
      }
#pragma unroll
    for (int c = 0; c < 16; ++c) sRaw[tl * RAWSTR + gsel * 16 + c] = acc[c];
  }
  __syncthreads();

  // ---- phase 1: lig_x = nb@ligW + b ; rec_x = raw@recW + b (per wave, 16 rows) ----
  {
    v16bf aL = load_a_frag_nb(nb, row0 + rb, Npts);
    v16bf aR = load_a_frag_raw(sRaw, rb);
#pragma unroll
    for (int t = 0; t < 8; ++t) {
      v8f c = vzero8();
      c = wmma_bf16(aL, load_b_frag(sW, t * 16, 0), c);
      float bv = sBias[t * 16 + nidx];
#pragma unroll
      for (int r = 0; r < 8; ++r) c[r] += bv;
      store_d_bf(sLig, rb, t * 16, c);

      v8f d = vzero8();
      d = wmma_bf16(aR, load_b_frag(sW, 128 + t * 16, 0), d);
      float bv2 = sBias[128 + t * 16 + nidx];
#pragma unroll
      for (int r = 0; r < 8; ++r) d[r] += bv2;
      store_d_bf(sRec, rb, t * 16, d);
    }
  }
  __syncthreads();

  // ---- weight phase 2: lig_both^T (rows 0..127) + rec_both^T (rows 128..255) ----
  for (int i = tid; i < 128 * 128; i += THREADS) {
    int o = i & 127, k = i >> 7;
    sW[o * WSTR + k]         = f2bf(lbW[k * 128 + o]);
    sW[(128 + o) * WSTR + k] = f2bf(rbW[k * 128 + o]);
  }
  __syncthreads();

  // ---- phase 2: both = (lig_x@lbW + b) * (rec_x@rbW + b) ----
  {
    v16bf aL[4], aR[4];
#pragma unroll
    for (int ks = 0; ks < 4; ++ks) {
      aL[ks] = load_a_frag_bf(sLig, rb, ks * 32);
      aR[ks] = load_a_frag_bf(sRec, rb, ks * 32);
    }
#pragma unroll
    for (int t = 0; t < 8; ++t) {
      v8f cl = vzero8(), cr = vzero8();
#pragma unroll
      for (int ks = 0; ks < 4; ++ks) {
        cl = wmma_bf16(aL[ks], load_b_frag(sW, t * 16, ks * 32), cl);
        cr = wmma_bf16(aR[ks], load_b_frag(sW, 128 + t * 16, ks * 32), cr);
      }
      float bl = sBias[256 + t * 16 + nidx], br = sBias[384 + t * 16 + nidx];
#pragma unroll
      for (int r = 0; r < 8; ++r) cl[r] = (cl[r] + bl) * (cr[r] + br);
      store_d_bf(sBoth, rb, t * 16, cl);
    }
  }

  // ---- phase 3: final GEMM (K = 384 in three K=128 chunks: lig_x, rec_x, both) ----
  v8f facc[16];
#pragma unroll
  for (int t = 0; t < 16; ++t) facc[t] = vzero8();

  for (int c = 0; c < 3; ++c) {
    __syncthreads();  // prior readers of sW (and writers of sBoth) done
    for (int i = tid; i < 256 * 128; i += THREADS) {
      int o = i & 255, k = i >> 8;
      sW[o * WSTR + k] = f2bf(fW[(c * 128 + k) * 256 + o]);
    }
    __syncthreads();
    const unsigned short* stgp = (c == 0) ? sLig : ((c == 1) ? sRec : sBoth);
    v16bf a[4];
#pragma unroll
    for (int ks = 0; ks < 4; ++ks) a[ks] = load_a_frag_bf(stgp, rb, ks * 32);
#pragma unroll
    for (int t = 0; t < 16; ++t) {
#pragma unroll
      for (int ks = 0; ks < 4; ++ks)
        facc[t] = wmma_bf16(a[ks], load_b_frag(sW, t * 16, ks * 32), facc[t]);
    }
  }

  // ---- bias + store (guard ragged tail) ----
  {
    int mo = (lane & 16) ? 8 : 0;
#pragma unroll
    for (int t = 0; t < 16; ++t) {
      float bv = sBias[512 + t * 16 + nidx];
#pragma unroll
      for (int r = 0; r < 8; ++r) {
        int row = row0 + rb + mo + r;
        if (row < Npts) out[(size_t)row * 256 + t * 16 + nidx] = facc[t][r] + bv;
      }
    }
  }
}

extern "C" void kernel_launch(void* const* d_in, const int* in_sizes, int n_in,
                              void* d_out, int out_size, void* d_ws, size_t ws_size,
                              hipStream_t stream) {
  (void)n_in; (void)out_size; (void)d_ws; (void)ws_size;
  const float* pos  = (const float*)d_in[0];
  const float* nb   = (const float*)d_in[1];
  // d_in[2] = batch (unused by the reference output)
  const float* g0   = (const float*)d_in[3];
  const float* g1   = (const float*)d_in[4];
  const float* ligW = (const float*)d_in[5];
  const float* ligb = (const float*)d_in[6];
  const float* recW = (const float*)d_in[7];
  const float* recb = (const float*)d_in[8];
  const float* lbW  = (const float*)d_in[9];
  const float* lbb  = (const float*)d_in[10];
  const float* rbW  = (const float*)d_in[11];
  const float* rbb  = (const float*)d_in[12];
  const float* fW   = (const float*)d_in[13];
  const float* fb   = (const float*)d_in[14];
  float* out = (float*)d_out;

  int Npts = in_sizes[0] / 3;
  int blocks = (Npts + ROWS_PER_BLOCK - 1) / ROWS_PER_BLOCK;
  hipLaunchKernelGGL(gridnet_fused, dim3(blocks), dim3(THREADS), 0, stream,
                     pos, nb, g0, g1, ligW, ligb, recW, recb,
                     lbW, lbb, rbW, rbb, fW, fb, out, Npts);
}
